// AdvancedCADGNN_11965778886761
// MI455X (gfx1250) — compile-verified, matrix operand order checked by reading, not run on hardware
//
#include <hip/hip_runtime.h>
#include <hip/hip_bf16.h>
#include <cstdint>

typedef __attribute__((ext_vector_type(16))) __bf16 v16bf;
typedef __attribute__((ext_vector_type(8)))  float  v8f;

#define DEVFN __device__ __forceinline__

// ---------------------------------------------------------------- utilities

__global__ void k_fill(float* __restrict__ p, float v, long long n) {
  long long i = (long long)blockIdx.x * blockDim.x + threadIdx.x;
  if (i < n) p[i] = v;
}

DEVFN unsigned short f2bf(float f) {
  uint32_t x = __float_as_uint(f);
  return (unsigned short)((x + 0x7FFFu + ((x >> 16) & 1u)) >> 16);  // RNE
}

__global__ void k_f32_to_bf16(const float* __restrict__ in,
                              unsigned short* __restrict__ out, long long n) {
  long long i = (long long)blockIdx.x * blockDim.x + threadIdx.x;
  if (i < n) out[i] = f2bf(in[i]);
}

// weights: convert f32 [K x NN] -> bf16 transposed [NN x K] (K-major rows)
__global__ void k_f32_to_bf16_T(const float* __restrict__ in,
                                unsigned short* __restrict__ out, int K, int NN) {
  int t = blockIdx.x * blockDim.x + threadIdx.x;
  if (t >= K * NN) return;
  int n = t / K, k = t - n * K;
  out[(size_t)n * K + k] = f2bf(in[(size_t)k * NN + n]);
}

DEVFN void atomicMaxF(float* addr, float v) {
  // monotonic float encoding: works for mixed signs with -inf init
  if (v >= 0.f) atomicMax((int*)addr, __float_as_int(v));
  else          atomicMin((unsigned int*)addr, __float_as_uint(v));
}

// ------------------------------------------------- edge-attr mean (16 feats)

__global__ void k_edge_mean(const float* __restrict__ ea, float* __restrict__ mean,
                            int E, int FE) {
  __shared__ float red[256];
  int f = blockIdx.x, t = threadIdx.x;
  float s = 0.f;
  for (int e = t; e < E; e += 256) s += ea[(size_t)e * FE + f];
  red[t] = s; __syncthreads();
  for (int w = 128; w > 0; w >>= 1) { if (t < w) red[t] += red[t + w]; __syncthreads(); }
  if (t == 0) mean[f] = red[0] / (float)E;
}

// ------------------------------------- fold We @ a_e  ->  q[FE,H]  (tiny)

__global__ void k_fold(const float* __restrict__ W, const float* __restrict__ a,
                       float* __restrict__ q, int Fin, int H, int C) {
  int t = blockIdx.x * blockDim.x + threadIdx.x;
  if (t >= Fin * H) return;
  int f = t / H, h = t % H;
  float s = 0.f;
  for (int c = 0; c < C; ++c) s += W[(size_t)f * H * C + h * C + c] * a[h * C + c];
  q[f * H + h] = s;
}

__global__ void k_alpha_loop(const float* __restrict__ mean_ea, const float* __restrict__ qe,
                             float* __restrict__ aloop, int FE, int H) {
  int h = threadIdx.x;
  if (h >= H) return;
  float s = 0.f;
  for (int f = 0; f < FE; ++f) s += mean_ea[f] * qe[f * H + h];
  aloop[h] = s;
}

// ------------------------------------- per-node attention dots from xs

__global__ void k_node_alpha(const float* __restrict__ xs, const float* __restrict__ as_,
                             const float* __restrict__ ad_, float* __restrict__ oas,
                             float* __restrict__ oad, int N, int H, int C) {
  int t = blockIdx.x * blockDim.x + threadIdx.x;
  if (t >= N * H) return;
  int n = t / H, h = t % H;
  const float* row = xs + (size_t)n * H * C + (size_t)h * C;
  float s1 = 0.f, s2 = 0.f;
  for (int c = 0; c < C; ++c) { float v = row[c]; s1 += v * as_[h * C + c]; s2 += v * ad_[h * C + c]; }
  oas[t] = s1; oad[t] = s2;
}

// ------------------------------------- edge pass 1: logit + segment max

__global__ void k_edge_logit(const int* __restrict__ ei, const float* __restrict__ ea,
                             const float* __restrict__ qe, const float* __restrict__ aloop,
                             const float* __restrict__ asrc, const float* __restrict__ adst,
                             float* __restrict__ albuf, float* __restrict__ m,
                             int E, int N, int H) {
  int e = blockIdx.x * blockDim.x + threadIdx.x;
  if (e >= E + N) return;
  int s, d;
  float row[16];
  bool real = (e < E);
  if (real) {
    s = ei[e]; d = ei[E + e];
    const float4* r4 = (const float4*)&ea[(size_t)e * 16];
    #pragma unroll
    for (int f4 = 0; f4 < 4; ++f4) {
      float4 v = r4[f4];
      row[f4 * 4 + 0] = v.x; row[f4 * 4 + 1] = v.y;
      row[f4 * 4 + 2] = v.z; row[f4 * 4 + 3] = v.w;
    }
  } else {
    s = d = e - E;
  }
  for (int h = 0; h < H; ++h) {
    float aev;
    if (real) {
      aev = 0.f;
      #pragma unroll
      for (int f = 0; f < 16; ++f) aev += row[f] * qe[f * H + h];
    } else {
      aev = aloop[h];
    }
    float al = asrc[(size_t)s * H + h] + adst[(size_t)d * H + h] + aev;
    al = (al > 0.f) ? al : 0.2f * al;   // leaky relu
    albuf[(size_t)e * H + h] = al;
    atomicMaxF(&m[(size_t)d * H + h], al);
  }
}

// ------------------------------------- edge pass 2: exp + segment sum

__global__ void k_edge_exp(const int* __restrict__ ei, float* __restrict__ albuf,
                           const float* __restrict__ m, float* __restrict__ denom,
                           int E, int N, int H) {
  int e = blockIdx.x * blockDim.x + threadIdx.x;
  if (e >= E + N) return;
  int d = (e < E) ? ei[E + e] : (e - E);
  for (int h = 0; h < H; ++h) {
    float ex = __expf(albuf[(size_t)e * H + h] - m[(size_t)d * H + h]);
    albuf[(size_t)e * H + h] = ex;
    atomicAdd(&denom[(size_t)d * H + h], ex);
  }
}

// ------------------------------------- edge pass 3: weighted scatter-aggregate
// 4 channels / thread: float4 gather of xs[src] (L2-resident), 4 atomic adds

__global__ void k_aggregate4(const int* __restrict__ ei, const float* __restrict__ albuf,
                             const float* __restrict__ denom, const float* __restrict__ xs,
                             float* __restrict__ agg, int E, int N, int H,
                             int logHC4, int logC) {
  long long t = (long long)blockIdx.x * blockDim.x + threadIdx.x;
  long long total = (long long)(E + N) << logHC4;
  if (t >= total) return;
  int e  = (int)(t >> logHC4);
  int j4 = (int)(t & ((1 << logHC4) - 1)) << 2;
  int h  = j4 >> logC;
  int HC = 4 << logHC4;
  int s, d;
  if (e < E) { s = ei[e]; d = ei[E + e]; } else { s = d = e - E; }
  float w = albuf[(size_t)e * H + h] / (denom[(size_t)d * H + h] + 1e-16f);
  float4 v = *(const float4*)&xs[(size_t)s * HC + j4];
  float* o = &agg[(size_t)d * HC + j4];
  atomicAdd(o + 0, w * v.x);
  atomicAdd(o + 1, w * v.y);
  atomicAdd(o + 2, w * v.z);
  atomicAdd(o + 3, w * v.w);
}

// ------------------------------------- bias + elu

__global__ void k_bias_elu(const float* __restrict__ agg, const float* __restrict__ b,
                           float* __restrict__ out, long long n, int HC) {
  long long t = (long long)blockIdx.x * blockDim.x + threadIdx.x;
  if (t >= n) return;
  int j = (int)(t % HC);
  float v = agg[t] + b[j];
  out[t] = (v > 0.f) ? v : (__expf(v) - 1.f);
}

// ------------------------------------- bias + log_softmax (NC = 16)

__global__ void k_final(const float* __restrict__ agg, const float* __restrict__ b,
                        float* __restrict__ out, int N) {
  int n = blockIdx.x * blockDim.x + threadIdx.x;
  if (n >= N) return;
  float v[16]; float mx = -3.0e38f;
  #pragma unroll
  for (int j = 0; j < 16; ++j) { v[j] = agg[(size_t)n * 16 + j] + b[j]; mx = fmaxf(mx, v[j]); }
  float s = 0.f;
  #pragma unroll
  for (int j = 0; j < 16; ++j) s += __expf(v[j] - mx);
  float l = logf(s) + mx;
  #pragma unroll
  for (int j = 0; j < 16; ++j) out[(size_t)n * 16 + j] = v[j] - l;
}

// ------------------------------------- WMMA bf16 tiled GEMM, double-buffered
// C[M x NN] = A[M x K] * B (B supplied TRANSPOSED: BT[NN x K], K-major).
// block = 256 threads = 8 waves, wave tile WTM x WTN. LDS A row-major and B
// fragment-major are BOTH contiguous in global memory -> fills are one
// global_load_b128 + one ds_store_b128 per 16B chunk. Two LDS buffers:
// ONE barrier per K-step, next tile's loads overlap current tile's wmmas.
// FULL template param: hot blocks compile with zero predication; the single
// M-edge block gets its own launch (mOff rebases blockIdx.x).

template <int MT, int NT, int WTM, int WTN, int NN, bool FULL>
__global__ void k_gemm_bf16(const unsigned short* __restrict__ A,
                            const unsigned short* __restrict__ BT,
                            float* __restrict__ C, int M, int K, int mOff) {
  constexpr int WM = MT / WTM, WN = NT / WTN;
  constexpr int NACCM = WTM / 16, NACCN = WTN / 16;
  static_assert(WM * WN == 8, "8 waves per block");
  __shared__ unsigned short lA[2][MT * 32];
  __shared__ unsigned short lB[2][NT * 32];
  const int tid  = threadIdx.x;
  const int lane = tid & 31;
  const int w    = tid >> 5;
  const int wm = w / WN, wn = w % WN;
  const int bM = (mOff + blockIdx.x) * MT, bN = blockIdx.y * NT;
  const int r = lane & 15, half = lane >> 4;

  auto fill = [&](int buf, int k0) {
    #pragma unroll
    for (int i = tid; i < MT * 4; i += 256) {
      int row = i >> 2, c4 = i & 3;
      const unsigned short* src = &A[(size_t)(bM + row) * K + k0 + c4 * 8];
      uint4 q;
      if (FULL) {
        q = *(const uint4*)src;
        __builtin_prefetch(src + 64, 0, 1);   // next-next K-tile (speculative)
      } else {
        q = (uint4){0u, 0u, 0u, 0u};
        if (bM + row < M) q = *(const uint4*)src;
      }
      *(uint4*)&lA[buf][row * 32 + c4 * 8] = q;
    }
    #pragma unroll
    for (int i = tid; i < NT * 4; i += 256) {
      int col = i >> 2, c4 = i & 3;
      const unsigned short* src = &BT[(size_t)(bN + col) * K + k0 + c4 * 8];
      *(uint4*)&lB[buf][col * 32 + c4 * 8] = *(const uint4*)src;
      if (FULL) __builtin_prefetch(src + 64, 0, 1);
    }
  };

  v8f acc[NACCM][NACCN];
  #pragma unroll
  for (int im = 0; im < NACCM; ++im)
    #pragma unroll
    for (int in = 0; in < NACCN; ++in) acc[im][in] = (v8f){};

  fill(0, 0);                                 // prologue
  int cur = 0;
  for (int k0 = 0; k0 < K; k0 += 32, cur ^= 1) {
    __syncthreads();                          // fill(cur) visible; buf[cur^1] free
    if (k0 + 32 < K) fill(cur ^ 1, k0 + 32);  // overlap with wmmas below
    // ---- fragments: two contiguous 16B LDS chunks each ----
    v16bf af[NACCM], bf[NACCN];
    #pragma unroll
    for (int im = 0; im < NACCM; ++im) {
      const unsigned short* pa = &lA[cur][(wm * WTM + im * 16 + r) * 32 + 8 * half];
      ((uint4*)&af[im])[0] = *(const uint4*)pa;
      ((uint4*)&af[im])[1] = *(const uint4*)(pa + 16);
    }
    #pragma unroll
    for (int in = 0; in < NACCN; ++in) {
      const unsigned short* pb = &lB[cur][(wn * WTN + in * 16 + r) * 32 + 8 * half];
      ((uint4*)&bf[in])[0] = *(const uint4*)pb;
      ((uint4*)&bf[in])[1] = *(const uint4*)(pb + 16);
    }
    #pragma unroll
    for (int im = 0; im < NACCM; ++im)
      #pragma unroll
      for (int in = 0; in < NACCN; ++in)
        acc[im][in] = __builtin_amdgcn_wmma_f32_16x16x32_bf16(
            false, af[im], false, bf[in], (short)0, acc[im][in], false, false);
  }
  // ---- store C: D layout -> VGPR i holds row (i + 8*half), col = lane&15 ----
  #pragma unroll
  for (int im = 0; im < NACCM; ++im) {
    const int row0 = bM + wm * WTM + im * 16 + 8 * half;
    #pragma unroll
    for (int in = 0; in < NACCN; ++in) {
      const int col0 = bN + wn * WTN + in * 16 + (lane & 15);
      if (FULL) {
        float* base = &C[(size_t)row0 * NN + col0];
        #pragma unroll
        for (int i = 0; i < 8; ++i) base[(size_t)i * NN] = acc[im][in][i];
      } else {
        #pragma unroll
        for (int i = 0; i < 8; ++i) {
          int row = row0 + i;
          if (row < M) C[(size_t)row * NN + col0] = acc[im][in][i];
        }
      }
    }
  }
}

// ---------------------------------------------------------------- launcher

extern "C" void kernel_launch(void* const* d_in, const int* in_sizes, int n_in,
                              void* d_out, int out_size, void* d_ws, size_t ws_size,
                              hipStream_t stream) {
  (void)in_sizes; (void)n_in; (void)out_size; (void)ws_size;
  const int N = 20000, E = 400000, FIN = 128, FE = 16, NC = 16, HID = 128, HH = 4;
  const int Et = E + N;

  const float* x   = (const float*)d_in[0];
  const int*   ei  = (const int*)  d_in[1];
  const float* ea  = (const float*)d_in[2];
  const float* W1  = (const float*)d_in[3];
  const float* We1 = (const float*)d_in[4];
  const float* a1s = (const float*)d_in[5];
  const float* a1d = (const float*)d_in[6];
  const float* a1e = (const float*)d_in[7];
  const float* b1  = (const float*)d_in[8];
  const float* W2  = (const float*)d_in[9];
  const float* We2 = (const float*)d_in[10];
  const float* a2s = (const float*)d_in[11];
  const float* a2d = (const float*)d_in[12];
  const float* a2e = (const float*)d_in[13];
  const float* b2  = (const float*)d_in[14];
  const float* W3  = (const float*)d_in[15];
  const float* We3 = (const float*)d_in[16];
  const float* a3s = (const float*)d_in[17];
  const float* a3d = (const float*)d_in[18];
  const float* a3e = (const float*)d_in[19];
  const float* b3  = (const float*)d_in[20];
  float* out = (float*)d_out;

  char* p = (char*)d_ws;
  auto carve = [&](size_t bytes) { char* r = p; p += (bytes + 255) & ~(size_t)255; return r; };
  float*          mean_ea = (float*)carve(16 * 4);
  float*          qe      = (float*)carve((size_t)FE * HH * 4);
  float*          aloop   = (float*)carve(HH * 4);
  unsigned short* Abf     = (unsigned short*)carve((size_t)N * 512 * 2);
  unsigned short* WbfT    = (unsigned short*)carve((size_t)512 * 512 * 2);
  float*          xs      = (float*)carve((size_t)N * 512 * 4);
  float*          agg     = (float*)carve((size_t)N * 512 * 4);
  float*          hcur    = (float*)carve((size_t)N * 512 * 4);
  float*          asrc    = (float*)carve((size_t)N * HH * 4);
  float*          adst    = (float*)carve((size_t)N * HH * 4);
  float*          mbuf    = (float*)carve((size_t)N * HH * 4);
  float*          dbuf    = (float*)carve((size_t)N * HH * 4);
  float*          albuf   = (float*)carve((size_t)Et * HH * 4);

  const int B = 256;
  auto nb = [&](long long n) { return (unsigned)((n + B - 1) / B); };

  const int MB_FULL = N / 128;                 // 156 full 128-row blocks
  const int MB_EDGE = (N % 128) ? 1 : 0;       // 1 edge block

  k_edge_mean<<<FE, B, 0, stream>>>(ea, mean_ea, E, FE);

  auto run_layer = [&](const float* Xin, int Fin, const float* W, const float* We,
                       const float* av_s, const float* av_d, const float* av_e,
                       int Hh, int C, int logHC4, int logC, bool wide) {
    const int HC = Hh * C;
    k_f32_to_bf16<<<nb((long long)N * Fin), B, 0, stream>>>(Xin, Abf, (long long)N * Fin);
    k_f32_to_bf16_T<<<nb((long long)Fin * HC), B, 0, stream>>>(W, WbfT, Fin, HC);
    if (wide) {
      k_gemm_bf16<128, 64, 32, 32, 512, true>
          <<<dim3(MB_FULL, 8), B, 0, stream>>>(Abf, WbfT, xs, N, Fin, 0);
      if (MB_EDGE)
        k_gemm_bf16<128, 64, 32, 32, 512, false>
            <<<dim3(MB_EDGE, 8), B, 0, stream>>>(Abf, WbfT, xs, N, Fin, MB_FULL);
    } else {
      k_gemm_bf16<128, 16, 16, 16, 16, true>
          <<<dim3(MB_FULL, 1), B, 0, stream>>>(Abf, WbfT, xs, N, Fin, 0);
      if (MB_EDGE)
        k_gemm_bf16<128, 16, 16, 16, 16, false>
            <<<dim3(MB_EDGE, 1), B, 0, stream>>>(Abf, WbfT, xs, N, Fin, MB_FULL);
    }
    k_fold<<<nb(FE * Hh), B, 0, stream>>>(We, av_e, qe, FE, Hh, C);
    k_alpha_loop<<<1, 32, 0, stream>>>(mean_ea, qe, aloop, FE, Hh);
    k_node_alpha<<<nb((long long)N * Hh), B, 0, stream>>>(xs, av_s, av_d, asrc, adst, N, Hh, C);
    k_fill<<<nb((long long)N * Hh), B, 0, stream>>>(mbuf, -3.0e38f, (long long)N * Hh);
    k_fill<<<nb((long long)N * Hh), B, 0, stream>>>(dbuf, 0.f, (long long)N * Hh);
    k_edge_logit<<<nb(Et), B, 0, stream>>>(ei, ea, qe, aloop, asrc, adst, albuf, mbuf, E, N, Hh);
    k_edge_exp<<<nb(Et), B, 0, stream>>>(ei, albuf, mbuf, dbuf, E, N, Hh);
    k_fill<<<nb((long long)N * HC), B, 0, stream>>>(agg, 0.f, (long long)N * HC);
    k_aggregate4<<<nb(((long long)Et) << logHC4), B, 0, stream>>>(ei, albuf, dbuf, xs, agg,
                                                                  E, N, Hh, logHC4, logC);
  };

  // layer 1: 128 -> 4x128 concat
  run_layer(x, FIN, W1, We1, a1s, a1d, a1e, HH, HID, 7, 7, true);
  k_bias_elu<<<nb((long long)N * 512), B, 0, stream>>>(agg, b1, hcur, (long long)N * 512, 512);
  // layer 2: 512 -> 4x128 concat
  run_layer(hcur, 512, W2, We2, a2s, a2d, a2e, HH, HID, 7, 7, true);
  k_bias_elu<<<nb((long long)N * 512), B, 0, stream>>>(agg, b2, hcur, (long long)N * 512, 512);
  // layer 3: 512 -> 16, single head
  run_layer(hcur, 512, W3, We3, a3s, a3d, a3e, 1, NC, 2, 4, false);
  k_final<<<nb(N), B, 0, stream>>>(agg, b3, out, N);
}
